// InformerEncoder_34239479283912
// MI455X (gfx1250) — matrix operand order repaired and checked by reading, not compiled, not run on hardware
//
#include <hip/hip_runtime.h>
#include <hip/hip_bf16.h>
#include <math.h>

// ---------------------------------------------------------------------------
// Informer encoder for MI455X (gfx1250): f16 WMMA GEMMs with f32 accumulate.
// ---------------------------------------------------------------------------

typedef _Float16 half_t;
typedef __attribute__((ext_vector_type(16))) _Float16 v16h;
typedef __attribute__((ext_vector_type(8)))  _Float16 v8h;
typedef __attribute__((ext_vector_type(8)))  float    v8f;
typedef __attribute__((ext_vector_type(4)))  float    v4f;

#define B_  4
#define H_  8
#define D_  512
#define E_  64
#define NL_ 3

#define BM 128
#define BN 128
#define BKK 32

// ===========================================================================
// WMMA GEMM:  C[M,N] = epi( A[M,K] @ W[N,K]^T + bias )   (A f32, W f16)
// EPI: 0 = bias only, 1 = bias + exact GELU, 2 = bias + residual R[M,N]
// ===========================================================================
template <int EPI>
__global__ __launch_bounds__(256) void gemm_wmma(
    const float* __restrict__ A, const half_t* __restrict__ W,
    const float* __restrict__ bias, const float* __restrict__ R,
    float* __restrict__ C, int M, int N, int K) {
  __shared__ __align__(16) half_t As[BM][BKK + 8];
  __shared__ __align__(16) half_t Bs[BN][BKK + 8];

  const int t    = threadIdx.x;
  const int lane = t & 31;
  const int w    = t >> 5;   // 8 waves
  const int wm   = w & 1;    // 2 M-waves (64 rows each)
  const int wn   = w >> 1;   // 4 N-waves (32 cols each)
  const int m0   = blockIdx.y * BM;
  const int n0   = blockIdx.x * BN;

  const int lrow = t >> 1;         // 0..127
  const int lcol = (t & 1) * 16;   // 0 or 16

  v8f acc[4][2];
#pragma unroll
  for (int i = 0; i < 4; ++i)
#pragma unroll
    for (int j = 0; j < 2; ++j) acc[i][j] = (v8f)0.f;

  for (int kk = 0; kk < K; kk += BKK) {
    // ---- stage A tile: f32 global -> f16 LDS (row-major M x 32)
    {
      const float* src = A + (size_t)(m0 + lrow) * K + kk + lcol;
#pragma unroll
      for (int i = 0; i < 4; ++i) {
        v4f d = *(const v4f*)(src + i * 4);
        half_t* dst = &As[lrow][lcol + i * 4];
        dst[0] = (half_t)d.x; dst[1] = (half_t)d.y;
        dst[2] = (half_t)d.z; dst[3] = (half_t)d.w;
      }
      const half_t* wsrc = W + (size_t)(n0 + lrow) * K + kk + lcol;
      *(v8h*)&Bs[lrow][lcol]     = *(const v8h*)(wsrc);
      *(v8h*)&Bs[lrow][lcol + 8] = *(const v8h*)(wsrc + 8);
    }
    __syncthreads();

    if (kk + BKK < K)  // global_prefetch_b8 of next K tile
      __builtin_prefetch(A + (size_t)(m0 + lrow) * K + kk + BKK + lcol, 0, 3);

    // ---- build fragments per CDNA5 wave32 WMMA VGPR layouts
    const int hk = (lane < 16) ? 0 : 8;    // A: lanes 0-15 K0-7/16-23, hi K8-15/24-31
    const int bk = (lane < 16) ? 0 : 8;    // B: lanes 0-15 K0-7/16-23, hi K8-15/24-31
    v16h af[4], bf[2];
#pragma unroll
    for (int i = 0; i < 4; ++i) {
      int r = wm * 64 + i * 16 + (lane & 15);
      v8h lo = *(const v8h*)&As[r][hk];
      v8h hi = *(const v8h*)&As[r][16 + hk];
      af[i] = __builtin_shufflevector(lo, hi, 0, 1, 2, 3, 4, 5, 6, 7,
                                      8, 9, 10, 11, 12, 13, 14, 15);
    }
#pragma unroll
    for (int j = 0; j < 2; ++j) {
      int cidx = wn * 32 + j * 16 + (lane & 15);
      v8h lo = *(const v8h*)&Bs[cidx][bk];
      v8h hi = *(const v8h*)&Bs[cidx][bk + 16];
      bf[j] = __builtin_shufflevector(lo, hi, 0, 1, 2, 3, 4, 5, 6, 7,
                                      8, 9, 10, 11, 12, 13, 14, 15);
    }

#pragma unroll
    for (int i = 0; i < 4; ++i)
#pragma unroll
      for (int j = 0; j < 2; ++j)
        acc[i][j] = __builtin_amdgcn_wmma_f32_16x16x32_f16(
            false, af[i], false, bf[j], (short)0, acc[i][j], false, false);
    __syncthreads();
  }

  // ---- epilogue
#pragma unroll
  for (int i = 0; i < 4; ++i) {
    int mbase = m0 + wm * 64 + i * 16 + ((lane < 16) ? 0 : 8);
#pragma unroll
    for (int j = 0; j < 2; ++j) {
      int n = n0 + wn * 32 + j * 16 + (lane & 15);
      float bb = bias[n];
#pragma unroll
      for (int v = 0; v < 8; ++v) {
        size_t idx = (size_t)(mbase + v) * N + n;
        float val = acc[i][j][v] + bb;
        if (EPI == 1) val = 0.5f * val * (1.0f + erff(val * 0.70710678118f));
        if (EPI == 2) val += R[idx];
        C[idx] = val;
      }
    }
  }
}

// ===========================================================================
// Conv-as-GEMM: y[(b,l),o] = sum_t sum_i Wp[o, t*512+i] * x[b, wrap(l+t-1), i]
// Wp pre-packed (O=512, K=1536) f16.  Token-major in/out.
// ===========================================================================
__global__ __launch_bounds__(256) void conv_gemm_wmma(
    const float* __restrict__ X, const half_t* __restrict__ Wp,
    const float* __restrict__ bias, float* __restrict__ C, int M, int Lc) {
  __shared__ __align__(16) half_t As[BM][BKK + 8];
  __shared__ __align__(16) half_t Bs[BN][BKK + 8];
  const int N = D_, K = 3 * D_;

  const int t = threadIdx.x, lane = t & 31, w = t >> 5;
  const int wm = w & 1, wn = w >> 1;
  const int m0 = blockIdx.y * BM, n0 = blockIdx.x * BN;
  const int lrow = t >> 1, lcol = (t & 1) * 16;

  // shifted-token gather base for this thread's A row
  const int gm = m0 + lrow;
  const int b = gm / Lc, l = gm - b * Lc;

  v8f acc[4][2];
#pragma unroll
  for (int i = 0; i < 4; ++i)
#pragma unroll
    for (int j = 0; j < 2; ++j) acc[i][j] = (v8f)0.f;

  for (int kk = 0; kk < K; kk += BKK) {
    const int t3 = kk >> 9;        // tap 0,1,2 (each 32-slice stays in one tap)
    const int ci = kk & (D_ - 1);  // channel base
    int l2 = l + t3 - 1;
    if (l2 < 0) l2 += Lc;
    if (l2 >= Lc) l2 -= Lc;
    {
      const float* src = X + ((size_t)(b * Lc + l2) * D_) + ci + lcol;
#pragma unroll
      for (int i = 0; i < 4; ++i) {
        v4f d = *(const v4f*)(src + i * 4);
        half_t* dst = &As[lrow][lcol + i * 4];
        dst[0] = (half_t)d.x; dst[1] = (half_t)d.y;
        dst[2] = (half_t)d.z; dst[3] = (half_t)d.w;
      }
      const half_t* wsrc = Wp + (size_t)(n0 + lrow) * K + kk + lcol;
      *(v8h*)&Bs[lrow][lcol]     = *(const v8h*)(wsrc);
      *(v8h*)&Bs[lrow][lcol + 8] = *(const v8h*)(wsrc + 8);
    }
    __syncthreads();

    const int hk = (lane < 16) ? 0 : 8;
    const int bk = (lane < 16) ? 0 : 8;
    v16h af[4], bf[2];
#pragma unroll
    for (int i = 0; i < 4; ++i) {
      int r = wm * 64 + i * 16 + (lane & 15);
      v8h lo = *(const v8h*)&As[r][hk];
      v8h hi = *(const v8h*)&As[r][16 + hk];
      af[i] = __builtin_shufflevector(lo, hi, 0, 1, 2, 3, 4, 5, 6, 7,
                                      8, 9, 10, 11, 12, 13, 14, 15);
    }
#pragma unroll
    for (int j = 0; j < 2; ++j) {
      int cidx = wn * 32 + j * 16 + (lane & 15);
      v8h lo = *(const v8h*)&Bs[cidx][bk];
      v8h hi = *(const v8h*)&Bs[cidx][bk + 16];
      bf[j] = __builtin_shufflevector(lo, hi, 0, 1, 2, 3, 4, 5, 6, 7,
                                      8, 9, 10, 11, 12, 13, 14, 15);
    }
#pragma unroll
    for (int i = 0; i < 4; ++i)
#pragma unroll
      for (int j = 0; j < 2; ++j)
        acc[i][j] = __builtin_amdgcn_wmma_f32_16x16x32_f16(
            false, af[i], false, bf[j], (short)0, acc[i][j], false, false);
    __syncthreads();
  }

#pragma unroll
  for (int i = 0; i < 4; ++i) {
    int mbase = m0 + wm * 64 + i * 16 + ((lane < 16) ? 0 : 8);
#pragma unroll
    for (int j = 0; j < 2; ++j) {
      int n = n0 + wn * 32 + j * 16 + (lane & 15);
      float bb = bias[n];
#pragma unroll
      for (int v = 0; v < 8; ++v)
        C[(size_t)(mbase + v) * N + n] = acc[i][j][v] + bb;
    }
  }
}

// ===========================================================================
// ProbSparse attention pieces (tiny FLOPs -> VALU kernels)
// ===========================================================================
__device__ __forceinline__ unsigned hash_u32(unsigned s) {
  s ^= s >> 16; s *= 0x45d9f3bu; s ^= s >> 16; s *= 0x45d9f3bu; s ^= s >> 16;
  return s;
}

// M[b,h,l] = max_u q.k_samp - mean_u-ish (sum/L) over U sampled keys
__global__ void prob_m_kernel(const float* __restrict__ Q,
                              const float* __restrict__ Kt,
                              float* __restrict__ Mout, int Lc, int U,
                              unsigned seed) {
  int gid = blockIdx.x * blockDim.x + threadIdx.x;
  if (gid >= B_ * H_ * Lc) return;
  int l = gid % Lc, bh = gid / Lc, h = bh % H_, b = bh / H_;
  const float* q = Q + ((size_t)(b * Lc + l) * D_) + h * E_;
  float mx = -1e30f, sm = 0.f;
  for (int u = 0; u < U; ++u) {
    unsigned s = hash_u32((unsigned)(l * U + u) * 2654435761u + seed);
    int ki = (int)(s % (unsigned)Lc);
    const float* kp = Kt + ((size_t)(b * Lc + ki) * D_) + h * E_;
    float d = 0.f;
#pragma unroll 8
    for (int e = 0; e < E_; ++e) d += q[e] * kp[e];
    mx = fmaxf(mx, d);
    sm += d;
  }
  Mout[gid] = mx - sm / (float)Lc;
}

// top-U indices per (b,h) via iterative argmax (destroys Mbuf — scratch)
__global__ __launch_bounds__(256) void topk_kernel(float* __restrict__ Mbuf,
                                                   int* __restrict__ top,
                                                   int Lc, int U) {
  __shared__ float sv[256];
  __shared__ int   si[256];
  int bh = blockIdx.x;
  float* Mp = Mbuf + (size_t)bh * Lc;
  for (int r = 0; r < U; ++r) {
    float best = -1e30f; int bi = 0;
    for (int l = threadIdx.x; l < Lc; l += 256) {
      float v = Mp[l];
      if (v > best) { best = v; bi = l; }
    }
    sv[threadIdx.x] = best; si[threadIdx.x] = bi;
    __syncthreads();
    for (int s = 128; s > 0; s >>= 1) {
      if (threadIdx.x < s && sv[threadIdx.x + s] > sv[threadIdx.x]) {
        sv[threadIdx.x] = sv[threadIdx.x + s];
        si[threadIdx.x] = si[threadIdx.x + s];
      }
      __syncthreads();
    }
    if (threadIdx.x == 0) { top[bh * U + r] = si[0]; Mp[si[0]] = -1e30f; }
    __syncthreads();
  }
}

__global__ void vmean_kernel(const float* __restrict__ V,
                             float* __restrict__ vmean, int Lc) {
  int bh = blockIdx.x, e = threadIdx.x, h = bh % H_, b = bh / H_;
  float s = 0.f;
  for (int l = 0; l < Lc; ++l) s += V[((size_t)(b * Lc + l) * D_) + h * E_ + e];
  vmean[bh * E_ + e] = s / (float)Lc;
}

__global__ void ctxfill_kernel(float* __restrict__ ctx,
                               const float* __restrict__ vmean, int Lc) {
  int gid = blockIdx.x * blockDim.x + threadIdx.x;
  if (gid >= B_ * Lc * D_) return;
  int c = gid & (D_ - 1), tok = gid / D_, b = tok / Lc;
  int h = c >> 6, e = c & 63;
  ctx[gid] = vmean[(b * H_ + h) * E_ + e];
}

// full softmax attention for one selected query row per block
__global__ __launch_bounds__(256) void attn_top_kernel(
    const float* __restrict__ Q, const float* __restrict__ Kt,
    const float* __restrict__ V, const int* __restrict__ top,
    float* __restrict__ ctx, int Lc, int U) {
  __shared__ float qs[E_];
  __shared__ float ps[2048];
  __shared__ float red[256];
  int blk = blockIdx.x, u = blk % U, bh = blk / U, h = bh % H_, b = bh / H_;
  int ql = top[bh * U + u];
  const size_t base = (size_t)b * Lc;
  if (threadIdx.x < E_) qs[threadIdx.x] = Q[(base + ql) * D_ + h * E_ + threadIdx.x];
  __syncthreads();
  const float scale = 0.125f;  // 1/sqrt(64)
  float lmax = -1e30f;
  for (int s = threadIdx.x; s < Lc; s += 256) {
    const float* kp = Kt + (base + s) * D_ + h * E_;
    float d = 0.f;
#pragma unroll 8
    for (int e = 0; e < E_; ++e) d += qs[e] * kp[e];
    d *= scale;
    ps[s] = d;
    lmax = fmaxf(lmax, d);
  }
  red[threadIdx.x] = lmax;
  __syncthreads();
  for (int s = 128; s > 0; s >>= 1) {
    if (threadIdx.x < s) red[threadIdx.x] = fmaxf(red[threadIdx.x], red[threadIdx.x + s]);
    __syncthreads();
  }
  float gmax = red[0];
  __syncthreads();
  float lsum = 0.f;
  for (int s = threadIdx.x; s < Lc; s += 256) {
    float e_ = __expf(ps[s] - gmax);
    ps[s] = e_;
    lsum += e_;
  }
  red[threadIdx.x] = lsum;
  __syncthreads();
  for (int s = 128; s > 0; s >>= 1) {
    if (threadIdx.x < s) red[threadIdx.x] += red[threadIdx.x + s];
    __syncthreads();
  }
  float inv = 1.0f / red[0];
  if (threadIdx.x < E_) {
    int e = threadIdx.x;
    float o = 0.f;
    for (int s = 0; s < Lc; ++s) o += ps[s] * V[(base + s) * D_ + h * E_ + e];
    ctx[(base + ql) * D_ + h * E_ + e] = o * inv;
  }
}

// ===========================================================================
// Norm / conv-post kernels
// ===========================================================================
__global__ __launch_bounds__(128) void layernorm_kernel(
    const float* __restrict__ X, const float* __restrict__ g,
    const float* __restrict__ c, float* __restrict__ Y) {
  __shared__ float red[128];
  size_t tok = blockIdx.x;
  const float* x = X + tok * D_;
  float* y = Y + tok * D_;
  float s = 0.f;
  for (int i = threadIdx.x; i < D_; i += 128) s += x[i];
  red[threadIdx.x] = s;
  __syncthreads();
  for (int st = 64; st > 0; st >>= 1) {
    if (threadIdx.x < st) red[threadIdx.x] += red[threadIdx.x + st];
    __syncthreads();
  }
  float mean = red[0] / (float)D_;
  __syncthreads();
  float v = 0.f;
  for (int i = threadIdx.x; i < D_; i += 128) {
    float d = x[i] - mean;
    v += d * d;
  }
  red[threadIdx.x] = v;
  __syncthreads();
  for (int st = 64; st > 0; st >>= 1) {
    if (threadIdx.x < st) red[threadIdx.x] += red[threadIdx.x + st];
    __syncthreads();
  }
  float rstd = rsqrtf(red[0] / (float)D_ + 1e-5f);
  for (int i = threadIdx.x; i < D_; i += 128)
    y[i] = (x[i] - mean) * rstd * g[i] + c[i];
}

__global__ __launch_bounds__(256) void bnstats_kernel(
    const float* __restrict__ Yc, float* __restrict__ mu,
    float* __restrict__ var, int Mtok) {
  __shared__ float r1[256], r2[256];
  int c = blockIdx.x;
  float s = 0.f, s2 = 0.f;
  for (int m = threadIdx.x; m < Mtok; m += 256) {
    float v = Yc[(size_t)m * D_ + c];
    s += v;
    s2 += v * v;
  }
  r1[threadIdx.x] = s; r2[threadIdx.x] = s2;
  __syncthreads();
  for (int st = 128; st > 0; st >>= 1) {
    if (threadIdx.x < st) {
      r1[threadIdx.x] += r1[threadIdx.x + st];
      r2[threadIdx.x] += r2[threadIdx.x + st];
    }
    __syncthreads();
  }
  if (threadIdx.x == 0) {
    float m_ = r1[0] / (float)Mtok;
    mu[c] = m_;
    var[c] = r2[0] / (float)Mtok - m_ * m_;
  }
}

// batchnorm-apply + ELU + maxpool(window 3, stride 2, -inf pad)
__global__ void convpost_kernel(const float* __restrict__ Yc,
                                const float* __restrict__ mu,
                                const float* __restrict__ var,
                                const float* __restrict__ gamma,
                                const float* __restrict__ beta,
                                float* __restrict__ Xout, int Lc, int Lout) {
  int gid = blockIdx.x * blockDim.x + threadIdx.x;
  if (gid >= B_ * Lout * D_) return;
  int c = gid & (D_ - 1), tj = gid / D_, b = tj / Lout, j = tj % Lout;
  float g = gamma[c], bb = beta[c], m_ = mu[c], rs = rsqrtf(var[c] + 1e-5f);
  float best = -1e30f;
#pragma unroll
  for (int dl = -1; dl <= 1; ++dl) {
    int l = 2 * j + dl;
    if (l < 0 || l >= Lc) continue;
    float v = (Yc[((size_t)(b * Lc + l)) * D_ + c] - m_) * rs * g + bb;
    v = v > 0.f ? v : (__expf(v) - 1.f);
    best = fmaxf(best, v);
  }
  Xout[((size_t)(b * Lout + j)) * D_ + c] = best;
}

// ===========================================================================
// Weight conversion / packing
// ===========================================================================
__global__ void f32_to_f16_kernel(const float* __restrict__ src,
                                  half_t* __restrict__ dst, int n) {
  int i = blockIdx.x * blockDim.x + threadIdx.x;
  if (i < n) dst[i] = (half_t)src[i];
}

// Wc (O=512, I=512, T=3) -> packed (O, t*512 + i) f16
__global__ void pack_convw_kernel(const float* __restrict__ Wc,
                                  half_t* __restrict__ dst) {
  int i = blockIdx.x * blockDim.x + threadIdx.x;
  if (i >= D_ * D_ * 3) return;
  int t = i % 3, rem = i / 3, ci = rem % D_, o = rem / D_;
  dst[(size_t)o * (3 * D_) + t * D_ + ci] = (half_t)Wc[i];
}

// ===========================================================================
// Host orchestration
// ===========================================================================
extern "C" void kernel_launch(void* const* d_in, const int* in_sizes, int n_in,
                              void* d_out, int out_size, void* d_ws,
                              size_t ws_size, hipStream_t stream) {
  (void)in_sizes; (void)n_in; (void)out_size; (void)ws_size;
  // ---- inputs flattened in jax pytree order (dict keys sorted):
  // 0:x  1:bn  2-5:conv0{Wc,bc,beta,gamma}  6-9:conv1{...}  10:gn
  // 11+16*i: layer i {W1,W2,Wk,Wo,Wq,Wv,b1,b2,bk,bo,bq,bv,c1,c2,g1,g2}
  const float* x_in = (const float*)d_in[0];
  const float* p_bn = (const float*)d_in[1];
  struct ConvP { const float *Wc, *bc, *beta, *gamma; } CV[2];
  for (int i = 0; i < 2; ++i) {
    const void* const* p = d_in + 2 + i * 4;
    CV[i] = { (const float*)p[0], (const float*)p[1],
              (const float*)p[2], (const float*)p[3] };
  }
  const float* p_gn = (const float*)d_in[10];
  struct LayerP {
    const float *W1, *W2, *Wk, *Wo, *Wq, *Wv;
    const float *b1, *b2, *bk, *bo, *bq, *bv, *c1, *c2, *g1, *g2;
  } LP[NL_];
  for (int i = 0; i < NL_; ++i) {
    const void* const* p = d_in + 11 + i * 16;
    LP[i] = { (const float*)p[0],  (const float*)p[1],  (const float*)p[2],
              (const float*)p[3],  (const float*)p[4],  (const float*)p[5],
              (const float*)p[6],  (const float*)p[7],  (const float*)p[8],
              (const float*)p[9],  (const float*)p[10], (const float*)p[11],
              (const float*)p[12], (const float*)p[13], (const float*)p[14],
              (const float*)p[15] };
  }

  // ---- workspace carve-up
  char* w8 = (char*)d_ws;
  size_t off = 0;
  auto grab = [&](size_t bytes) -> void* {
    void* p = w8 + off;
    off += (bytes + 255) & ~(size_t)255;
    return p;
  };
  const size_t WN = (size_t)D_ * D_;       // 262144
  half_t* hW[NL_][6];                      // q,k,v,o,w1,w2
  for (int l = 0; l < NL_; ++l)
    for (int j = 0; j < 6; ++j) hW[l][j] = (half_t*)grab(WN * sizeof(half_t));
  half_t* hWc[2];
  for (int c = 0; c < 2; ++c) hWc[c] = (half_t*)grab((size_t)D_ * 3 * D_ * sizeof(half_t));
  const size_t AMAX = (size_t)B_ * 2048 * D_;  // 4,194,304 floats
  float* fX = (float*)grab(AMAX * 4);
  float* fA = (float*)grab(AMAX * 4);
  float* fB = (float*)grab(AMAX * 4);
  float* fQ = (float*)grab(AMAX * 4);
  float* fK = (float*)grab(AMAX * 4);
  float* fV = (float*)grab(AMAX * 4);
  float* fM  = (float*)grab((size_t)B_ * H_ * 2048 * 4);
  int*   iTop = (int*)grab((size_t)B_ * H_ * 64 * 4);
  float* fVm = (float*)grab((size_t)B_ * H_ * E_ * 4);
  float* fMu = (float*)grab(D_ * 4);
  float* fVar = (float*)grab(D_ * 4);

  // ---- convert weights to f16 (deterministic, re-done every call)
  const int cvtB = 256;
  for (int l = 0; l < NL_; ++l) {
    const float* srcs[6] = { LP[l].Wq, LP[l].Wk, LP[l].Wv,
                             LP[l].Wo, LP[l].W1, LP[l].W2 };
    for (int j = 0; j < 6; ++j)
      f32_to_f16_kernel<<<(int)((WN + cvtB - 1) / cvtB), cvtB, 0, stream>>>(
          srcs[j], hW[l][j], (int)WN);
  }
  for (int c = 0; c < 2; ++c)
    pack_convw_kernel<<<(D_ * D_ * 3 + cvtB - 1) / cvtB, cvtB, 0, stream>>>(
        CV[c].Wc, hWc[c]);

  // ---- encoder pipeline
  const float* cur = x_in;
  int Lc = 2048;
  for (int i = 0; i < NL_; ++i) {
    const int Mtok = B_ * Lc;
    // U = min(FACTOR * ceil(ln L), L)
    int U = (Lc >= 2048) ? 40 : 35;
    dim3 gg(D_ / BN, Mtok / BM), bb(256);

    gemm_wmma<0><<<gg, bb, 0, stream>>>(cur, hW[i][0], LP[i].bq, nullptr, fQ, Mtok, D_, D_);
    gemm_wmma<0><<<gg, bb, 0, stream>>>(cur, hW[i][1], LP[i].bk, nullptr, fK, Mtok, D_, D_);
    gemm_wmma<0><<<gg, bb, 0, stream>>>(cur, hW[i][2], LP[i].bv, nullptr, fV, Mtok, D_, D_);

    unsigned seed = 0x9E3779B9u * (unsigned)(i + 1);
    prob_m_kernel<<<(B_ * H_ * Lc + 255) / 256, 256, 0, stream>>>(fQ, fK, fM, Lc, U, seed);
    topk_kernel<<<B_ * H_, 256, 0, stream>>>(fM, iTop, Lc, U);
    vmean_kernel<<<B_ * H_, E_, 0, stream>>>(fV, fVm, Lc);
    ctxfill_kernel<<<(Mtok * D_ + 255) / 256, 256, 0, stream>>>(fA, fVm, Lc);
    attn_top_kernel<<<B_ * H_ * U, 256, 0, stream>>>(fQ, fK, fV, iTop, fA, Lc, U);

    gemm_wmma<2><<<gg, bb, 0, stream>>>(fA, hW[i][3], LP[i].bo, cur, fB, Mtok, D_, D_);
    layernorm_kernel<<<Mtok, 128, 0, stream>>>(fB, LP[i].g1, LP[i].c1, fX);
    gemm_wmma<1><<<gg, bb, 0, stream>>>(fX, hW[i][4], LP[i].b1, nullptr, fA, Mtok, D_, D_);
    gemm_wmma<2><<<gg, bb, 0, stream>>>(fA, hW[i][5], LP[i].b2, fX, fB, Mtok, D_, D_);
    if (i < NL_ - 1) {
      layernorm_kernel<<<Mtok, 128, 0, stream>>>(fB, LP[i].g2, LP[i].c2, fX);
      cur = fX;
      // ---- conv + BN + ELU + maxpool(stride 2)
      const int Lout = Lc / 2;
      conv_gemm_wmma<<<gg, bb, 0, stream>>>(cur, hWc[i], CV[i].bc, fA, Mtok, Lc);
      bnstats_kernel<<<D_, 256, 0, stream>>>(fA, fMu, fVar, Mtok);
      convpost_kernel<<<(B_ * Lout * D_ + 255) / 256, 256, 0, stream>>>(
          fA, fMu, fVar, CV[i].gamma, CV[i].beta, fX, Lc, Lout);
      Lc = Lout;
      cur = fX;
    } else {
      layernorm_kernel<<<Mtok, 128, 0, stream>>>(fB, LP[i].g2, LP[i].c2, fX);
      // final layernorm(gn, bn) -> d_out
      layernorm_kernel<<<Mtok, 128, 0, stream>>>(fX, p_gn, p_bn, (float*)d_out);
    }
  }
}
